// MultiHeadSelfAttention_46943992545963
// MI455X (gfx1250) — matrix operand order repaired and checked by reading, not compiled, not run on hardware
//
#include <hip/hip_runtime.h>
#include <hip/hip_bf16.h>

#define EMBED_DIM 1024
#define NUM_HEADS 16
#define HEAD_DIM  64
#define BATCH     4
#define SEQ       2048

typedef __bf16 bf16_t;
typedef __attribute__((ext_vector_type(16))) __bf16 v16bf;
typedef __attribute__((ext_vector_type(8)))  __bf16 v8bf;
typedef __attribute__((ext_vector_type(4)))  __bf16 v4bf;
typedef __attribute__((ext_vector_type(8)))  float  v8f;
typedef __attribute__((ext_vector_type(4)))  float  v4f;

// ---------------------------------------------------------------------------
// WMMA wrapper: D = A(16x32 bf16) * B(32x16 bf16) + C(16x16 f32)
// RA/RB = operand-reuse hints (VOP3P OPSEL bits): set on an instruction whose
// A/B operand is identical to the previous identical instruction's operand.
// ---------------------------------------------------------------------------
template <bool RA, bool RB>
__device__ __forceinline__ v8f wmma_bf16(v16bf a, v16bf b, v8f c) {
  return __builtin_amdgcn_wmma_f32_16x16x32_bf16(
      /*neg_a=*/false, a, /*neg_b=*/false, b,
      /*c_mod=*/(short)0, c, RA, RB);
}

// A fragment (16 rows x K=32, row-major source, leading dim `ld` elems).
// lanes 0-15: row M=lane, K={0..7,16..23}; lanes 16-31: K={8..15,24..31}.
__device__ __forceinline__ v16bf load_frag_a(const bf16_t* p, int ld) {
  const int lane = threadIdx.x & 31;
  const int sub  = lane & 15;
  const int hi   = (lane >> 4) & 1;
  const bf16_t* q = p + (size_t)sub * ld + (hi ? 8 : 0);
  union { v16bf v; v8bf h[2]; } u;
  u.h[0] = *(const v8bf*)(q);
  u.h[1] = *(const v8bf*)(q + 16);
  return u.v;
}

// B fragment (K=32 x 16 cols) from row-major Bt[n][k] (B transposed), ld elems.
// lanes 0-15: col N=lane, K={0..7,16..23}; lanes 16-31: K={8..15,24..31}.
__device__ __forceinline__ v16bf load_frag_b(const bf16_t* p, int ld) {
  const int lane = threadIdx.x & 31;
  const int sub  = lane & 15;
  const int hi   = (lane >> 4) & 1;
  const bf16_t* q = p + (size_t)sub * ld + (hi ? 8 : 0);
  union { v16bf v; v8bf h[2]; } u;
  u.h[0] = *(const v8bf*)(q);
  u.h[1] = *(const v8bf*)(q + 16);
  return u.v;
}

// ---------------------------------------------------------------------------
// CDNA5 async copy: global -> LDS, 16 bytes per lane, tracked by ASYNCcnt.
// ---------------------------------------------------------------------------
__device__ __forceinline__ void async_load_b128(unsigned lds_off, const void* gaddr) {
  asm volatile("global_load_async_to_lds_b128 %0, %1, off"
               :: "v"(lds_off), "v"(gaddr) : "memory");
}
__device__ __forceinline__ void wait_asynccnt0() {
  asm volatile("s_wait_asynccnt 0" ::: "memory");
}

// Fixed-pattern xor lane swap via ds_swizzle (group-of-32: [14:10]=xor, [4:0]=and)
template <int MASK>
__device__ __forceinline__ float swz_xor(float x) {
  int r = __builtin_amdgcn_ds_swizzle(__builtin_bit_cast(int, x),
                                      (MASK << 10) | 0x1f);
  return __builtin_bit_cast(float, r);
}

// ---------------------------------------------------------------------------
// fp32 -> bf16 conversion (4 elems / thread)
// ---------------------------------------------------------------------------
__global__ __launch_bounds__(256) void cvt_f32_to_bf16(
    const float* __restrict__ in, bf16_t* __restrict__ out, int n) {
  int i = (blockIdx.x * 256 + threadIdx.x) * 4;
  if (i + 3 < n) {
    v4f x = *(const v4f*)(in + i);
    v4bf y;
    y[0] = (bf16_t)x[0]; y[1] = (bf16_t)x[1];
    y[2] = (bf16_t)x[2]; y[3] = (bf16_t)x[3];
    *(v4bf*)(out + i) = y;
  } else {
    for (int j = i; j < n; ++j) out[j] = (bf16_t)in[j];
  }
}

// ---------------------------------------------------------------------------
// GEMM: C[M x N] = A[M x K] * W[N x K]^T + bias[N]
// 256 threads = 8 waves. BM=256 (32 rows/wave as 2 M-frags), BN=64, BK=32.
// W tile async-copied to LDS (padded stride 40 bf16 = 80B, bank-conflict free).
// 8 WMMA per K-step per wave; B fragment shared by the M-frag pair (reuse_b).
// ---------------------------------------------------------------------------
template <bool BF16_OUT>
__global__ __launch_bounds__(256) void gemm_xwt_bias(
    const bf16_t* __restrict__ A, const bf16_t* __restrict__ W,
    const float* __restrict__ bias, void* __restrict__ Cv,
    int M, int N, int K) {
  __shared__ bf16_t Bs[64 * 40];

  const int lane = threadIdx.x & 31;
  const int wv   = threadIdx.x >> 5;
  const int sub  = lane & 15;
  const int hi   = (lane >> 4) & 1;
  const int tn   = blockIdx.x * 64;
  const int tm   = blockIdx.y * 256;

  const int srow = threadIdx.x >> 2;        // 0..63 (W row within tile)
  const int scol = (threadIdx.x & 3) * 8;   // 0,8,16,24
  const unsigned ldsB = (unsigned)(size_t)&Bs[srow * 40 + scol];
  const bf16_t* wrow = W + (size_t)(tn + srow) * K + scol;

  v8f acc0[4] = {}, acc1[4] = {};
  const bf16_t* arow0 = A + (size_t)(tm + wv * 32) * K;
  const bf16_t* arow1 = arow0 + (size_t)16 * K;

  for (int kt = 0; kt < K; kt += 32) {
    async_load_b128(ldsB, wrow + kt);                 // W tile -> LDS (ASYNCcnt)
    if (kt + 32 < K)
      __builtin_prefetch(wrow + kt + 32, 0, 1);       // global_prefetch_b8
    wait_asynccnt0();
    __syncthreads();

    v16bf a0 = load_frag_a(arow0 + kt, K);
    v16bf a1 = load_frag_a(arow1 + kt, K);
#pragma unroll
    for (int nt = 0; nt < 4; ++nt) {
      v16bf bfrag = load_frag_b(&Bs[nt * 16 * 40], 40);
      acc0[nt] = wmma_bf16<false, false>(a0, bfrag, acc0[nt]);
      acc1[nt] = wmma_bf16<false, true >(a1, bfrag, acc1[nt]);  // same B as prev
    }
    __syncthreads();
  }

#pragma unroll
  for (int mi = 0; mi < 2; ++mi) {
#pragma unroll
    for (int nt = 0; nt < 4; ++nt) {
      const int col = tn + nt * 16 + sub;
      const float bvv = bias[col];
      const v8f acc = mi ? acc1[nt] : acc0[nt];
#pragma unroll
      for (int r = 0; r < 8; ++r) {
        const int row = tm + wv * 32 + mi * 16 + r + hi * 8;
        const float val = acc[r] + bvv;
        if constexpr (BF16_OUT)
          ((bf16_t*)Cv)[(size_t)row * N + col] = (bf16_t)val;
        else
          ((float*)Cv)[(size_t)row * N + col] = val;
      }
    }
  }
}

// ---------------------------------------------------------------------------
// Flash attention. Q/K/V in (B,S,D) bf16, head h occupies cols [h*64, h*64+64).
// 256 threads = 8 waves, 16 query rows/wave -> 128 rows/block.
// Per 32-key tile: K tile async-copied to LDS (shared by all 8 waves),
// V tile transposed into LDS via packed dword scatter, probs re-shaped
// C-layout -> A-layout through per-wave LDS.
// ---------------------------------------------------------------------------
__global__ __launch_bounds__(256) void flash_attn(
    const bf16_t* __restrict__ Q, const bf16_t* __restrict__ Kmat,
    const bf16_t* __restrict__ V, const int* __restrict__ mask,
    bf16_t* __restrict__ O) {
  __shared__ bf16_t Ks[32 * 72];        // K tile row-major [key][hd], stride 72
  __shared__ bf16_t Vt[64 * 40];        // V^T tile [hd][key], stride 40
  __shared__ bf16_t Pl[8 * 16 * 40];    // per-wave probs 16x32, stride 40

  const int lane = threadIdx.x & 31;
  const int wv   = threadIdx.x >> 5;
  const int sub  = lane & 15;
  const int hi   = (lane >> 4) & 1;

  const int b  = blockIdx.y / NUM_HEADS;
  const int h  = blockIdx.y % NUM_HEADS;
  const int q0 = blockIdx.x * 128 + wv * 16;
  const int cb = h * HEAD_DIM;

  // Q A-fragments (rows q0..q0+15, hd dim split into two K=32 steps)
  const bf16_t* qbase = Q + ((size_t)b * SEQ + q0) * EMBED_DIM + cb;
  const v16bf qa0 = load_frag_a(qbase, EMBED_DIM);
  const v16bf qa1 = load_frag_a(qbase + 32, EMBED_DIM);

  v8f o0 = {}, o1 = {}, o2 = {}, o3 = {};
  float mrow[8], lrow[8];
#pragma unroll
  for (int r = 0; r < 8; ++r) { mrow[r] = -3.0e38f; lrow[r] = 0.0f; }

  bf16_t* pw = Pl + wv * (16 * 40);
  const float scale = 0.125f;   // 1/sqrt(64)

  // K async staging map: thread -> (key 0..31, hd chunk of 8)
  const int akey = threadIdx.x >> 3;
  const int ahd  = (threadIdx.x & 7) * 8;
  const unsigned ldsK = (unsigned)(size_t)&Ks[akey * 72 + ahd];
  const bf16_t* kg0 = Kmat + ((size_t)b * SEQ + akey) * EMBED_DIM + cb + ahd;

  // V transpose staging map: thread -> (key pair, 4 hd)
  const int vp2 = (threadIdx.x & 15) * 2;   // 0,2,..,30
  const int vg  = (threadIdx.x >> 4) * 4;   // 0,4,..,60
  const bf16_t* vg0 = V + ((size_t)b * SEQ + vp2) * EMBED_DIM + cb + vg;

  for (int kt = 0; kt < SEQ; kt += 32) {
    // --- stage K tile via async copy; V tile transposed via packed stores ---
    async_load_b128(ldsK, kg0 + (size_t)kt * EMBED_DIM);
    v4bf va = *(const v4bf*)(vg0 + (size_t)kt * EMBED_DIM);
    v4bf vb = *(const v4bf*)(vg0 + (size_t)(kt + 1) * EMBED_DIM);
#pragma unroll
    for (int j = 0; j < 4; ++j) {
      unsigned pk = (unsigned)__builtin_bit_cast(unsigned short, va[j]) |
                    ((unsigned)__builtin_bit_cast(unsigned short, vb[j]) << 16);
      *(unsigned*)&Vt[(vg + j) * 40 + vp2] = pk;
    }
    wait_asynccnt0();
    __syncthreads();

    // --- scores S = Q K^T (16 x 32), two N-tiles, K(hd)=64 as two steps ---
    v8f s0 = {}, s1 = {};
    s0 = wmma_bf16<false, false>(qa0, load_frag_b(&Ks[0], 72), s0);
    s0 = wmma_bf16<false, false>(qa1, load_frag_b(&Ks[32], 72), s0);
    s1 = wmma_bf16<false, false>(qa0, load_frag_b(&Ks[16 * 72], 72), s1);
    s1 = wmma_bf16<false, false>(qa1, load_frag_b(&Ks[16 * 72 + 32], 72), s1);

    const int mok0 = mask[(size_t)b * SEQ + kt + sub];
    const int mok1 = mask[(size_t)b * SEQ + kt + 16 + sub];

    // --- online softmax: row stats via half-wave ds_swizzle xor reductions ---
#pragma unroll
    for (int r = 0; r < 8; ++r) {
      float a = mok0 ? s0[r] * scale : -3.0e38f;
      float c = mok1 ? s1[r] * scale : -3.0e38f;
      float pm = fmaxf(a, c);
      pm = fmaxf(pm, swz_xor<1>(pm));
      pm = fmaxf(pm, swz_xor<2>(pm));
      pm = fmaxf(pm, swz_xor<4>(pm));
      pm = fmaxf(pm, swz_xor<8>(pm));
      const float mn = fmaxf(mrow[r], pm);
      const float alpha = __expf(mrow[r] - mn);
      mrow[r] = mn;
      const float e0 = __expf(a - mn);
      const float e1 = __expf(c - mn);
      float ps = e0 + e1;
      ps += swz_xor<1>(ps);
      ps += swz_xor<2>(ps);
      ps += swz_xor<4>(ps);
      ps += swz_xor<8>(ps);
      lrow[r] = lrow[r] * alpha + ps;
      o0[r] *= alpha; o1[r] *= alpha; o2[r] *= alpha; o3[r] *= alpha;
      const int prow = r + hi * 8;
      pw[prow * 40 + sub]      = (bf16_t)e0;
      pw[prow * 40 + 16 + sub] = (bf16_t)e1;
    }
    asm volatile("s_wait_dscnt 0" ::: "memory");  // LDS stores -> loads, same wave

    // --- O += P V : A = probs from LDS (shared across 4 WMMAs -> reuse_a) ---
    const v16bf pa = load_frag_a(pw, 40);
    o0 = wmma_bf16<false, false>(pa, load_frag_b(&Vt[0 * 16 * 40], 40), o0);
    o1 = wmma_bf16<true,  false>(pa, load_frag_b(&Vt[1 * 16 * 40], 40), o1);
    o2 = wmma_bf16<true,  false>(pa, load_frag_b(&Vt[2 * 16 * 40], 40), o2);
    o3 = wmma_bf16<true,  false>(pa, load_frag_b(&Vt[3 * 16 * 40], 40), o3);
    __syncthreads();
  }

  // --- normalize and store attended (bf16, (B,S,D) layout) ---
#pragma unroll
  for (int r = 0; r < 8; ++r) {
    const float inv = 1.0f / lrow[r];
    const int row = q0 + r + hi * 8;
    const size_t base = ((size_t)b * SEQ + row) * EMBED_DIM + cb;
    O[base + 0  + sub] = (bf16_t)(o0[r] * inv);
    O[base + 16 + sub] = (bf16_t)(o1[r] * inv);
    O[base + 32 + sub] = (bf16_t)(o2[r] * inv);
    O[base + 48 + sub] = (bf16_t)(o3[r] * inv);
  }
}

// ---------------------------------------------------------------------------
extern "C" void kernel_launch(void* const* d_in, const int* in_sizes, int n_in,
                              void* d_out, int out_size, void* d_ws, size_t ws_size,
                              hipStream_t stream) {
  const float* x    = (const float*)d_in[0];
  const int*   mask = (const int*)d_in[1];
  const float* Wq   = (const float*)d_in[2];
  const float* bq   = (const float*)d_in[3];
  const float* Wk   = (const float*)d_in[4];
  const float* bk   = (const float*)d_in[5];
  const float* Wv   = (const float*)d_in[6];
  const float* bv   = (const float*)d_in[7];
  const float* Wo   = (const float*)d_in[8];
  const float* bo   = (const float*)d_in[9];

  const int M = BATCH * SEQ;              // 8192
  const int D = EMBED_DIM;                // 1024
  const size_t XE = (size_t)M * D;        // 8,388,608 elems
  const size_t WE = (size_t)D * D;        // 1,048,576 elems

  bf16_t* ws  = (bf16_t*)d_ws;
  bf16_t* xb  = ws;
  bf16_t* wqb = xb  + XE;
  bf16_t* wkb = wqb + WE;
  bf16_t* wvb = wkb + WE;
  bf16_t* wob = wvb + WE;
  bf16_t* qb  = wob + WE;
  bf16_t* kb  = qb  + XE;
  bf16_t* vb  = kb  + XE;
  bf16_t* ab  = vb  + XE;

  // fp32 -> bf16 conversions
  cvt_f32_to_bf16<<<(int)(XE / 1024), 256, 0, stream>>>(x,  xb,  (int)XE);
  cvt_f32_to_bf16<<<(int)(WE / 1024), 256, 0, stream>>>(Wq, wqb, (int)WE);
  cvt_f32_to_bf16<<<(int)(WE / 1024), 256, 0, stream>>>(Wk, wkb, (int)WE);
  cvt_f32_to_bf16<<<(int)(WE / 1024), 256, 0, stream>>>(Wv, wvb, (int)WE);
  cvt_f32_to_bf16<<<(int)(WE / 1024), 256, 0, stream>>>(Wo, wob, (int)WE);

  // Q/K/V projections (bf16 out)
  dim3 gg(D / 64, M / 256);   // (16, 32)
  gemm_xwt_bias<true><<<gg, 256, 0, stream>>>(xb, wqb, bq, qb, M, D, D);
  gemm_xwt_bias<true><<<gg, 256, 0, stream>>>(xb, wkb, bk, kb, M, D, D);
  gemm_xwt_bias<true><<<gg, 256, 0, stream>>>(xb, wvb, bv, vb, M, D, D);

  // flash attention
  flash_attn<<<dim3(SEQ / 128, BATCH * NUM_HEADS), 256, 0, stream>>>(
      qb, kb, vb, mask, ab);

  // output projection (fp32 out)
  gemm_xwt_bias<false><<<gg, 256, 0, stream>>>(ab, wob, bo, d_out, M, D, D);
}
